// ChebyKANLayer_20401094656055
// MI455X (gfx1250) — compile-verified, matrix operand order checked by reading, not run on hardware
//
#include <hip/hip_runtime.h>
#include <hip/hip_bf16.h>

// ---------------------------------------------------------------------------
// ChebyKAN layer as a fused bf16 WMMA GEMM for gfx1250 (MI455X).
//   y[b,o] = bias[o] + sum_{i,d=1..8} T_d(tanh(x[b,i])) * C[i,o,d]
//   GEMM: M=8192, K=8192 (k = (d-1)*1024 + i), N=1024;  bias[o]=sum_i C[i,o,0]
// Stages (all on `stream`):
//   0) xt = tanh(x)                 f32 -> ws          (one-shot, L2 resident)
//   1) pack C[:, :, 1..8] -> Wb bf16, B-fragment-native layout Wb[kt][o][kk]
//   2) bias[o] = sum_i C[i,o,0]
//   3) fused GEMM: per 128x32 x-tile, write ALL 8 Chebyshev planes to LDS
//      (96 KB, CDNA5's 320KB WGP LDS makes this possible), then a barrier-free
//      64-WMMA read phase. Only 2 barriers per it-step.
// ---------------------------------------------------------------------------

#define I_DIM 1024
#define O_DIM 1024
#define BM 128
#define BN 256
#define NTHREADS 512
#define LDA 48            // padded LDS stride (bf16 elems), 96B = 16B multiple

typedef __attribute__((ext_vector_type(16))) __bf16 v16bf;
typedef __attribute__((ext_vector_type(8)))  float  v8f;
typedef __attribute__((ext_vector_type(2)))  __bf16 v2bf;

union FragBF { v16bf v; uint4 q[2]; };

__device__ __forceinline__ unsigned pack2bf(float a, float b) {
#if __has_builtin(__builtin_amdgcn_cvt_pk_bf16_f32)
    v2bf r = __builtin_amdgcn_cvt_pk_bf16_f32(a, b);
    return __builtin_bit_cast(unsigned, r);
#else
    v2bf r; r.x = (__bf16)a; r.y = (__bf16)b;
    return __builtin_bit_cast(unsigned, r);
#endif
}
__device__ __forceinline__ unsigned short f2bf(float a) {
    return __builtin_bit_cast(unsigned short, (__bf16)a);
}

// ---------------------------------------------------------------------------
// Stage 0: xt = tanh(x), elementwise, float4-vectorized.
// ---------------------------------------------------------------------------
__global__ __launch_bounds__(256) void tanh_kernel(
    const float4* __restrict__ X, float4* __restrict__ XT, int n4)
{
    int idx = blockIdx.x * 256 + threadIdx.x;
    if (idx < n4) {
        float4 v = X[idx];
        float4 r;
        r.x = tanhf(v.x);
        r.y = tanhf(v.y);
        r.z = tanhf(v.z);
        r.w = tanhf(v.w);
        XT[idx] = r;
    }
}

// ---------------------------------------------------------------------------
// Stage 1: pack C[i][o][d], d=1..8 (f32) -> Wb[kt][o][kk] (bf16),
// k = (d-1)*1024 + i, kt=k/32, kk=k%32. Grid: (I/32, O/32), 256 threads.
// ---------------------------------------------------------------------------
__global__ __launch_bounds__(256) void cheby_pack_w(
    const float* __restrict__ C, unsigned short* __restrict__ Wb)
{
    __shared__ unsigned short sh[8 * 32 * 33];
    const int i0 = blockIdx.x * 32;
    const int o0 = blockIdx.y * 32;
    const int t  = threadIdx.x;

    for (int idx = t; idx < 32 * 32 * 8; idx += 256) {
        int i   = idx >> 8;          // / (32*8)
        int rem = idx & 255;
        int o   = rem >> 3;
        int d   = (rem & 7) + 1;     // 1..8
        float v = C[((size_t)(i0 + i) * O_DIM + (o0 + o)) * 9 + d];
        sh[((d - 1) * 32 + o) * 33 + i] = f2bf(v);
    }
    __syncthreads();

    unsigned* W32 = (unsigned*)Wb;
    const int itile = i0 >> 5;
    for (int idx = t; idx < 8 * 32 * 16; idx += 256) {
        int dd  = idx >> 9;          // 0..7  == d-1
        int rem = idx & 511;
        int o   = rem >> 4;
        int kh  = rem & 15;
        unsigned lo = sh[(dd * 32 + o) * 33 + kh * 2];
        unsigned hi = sh[(dd * 32 + o) * 33 + kh * 2 + 1];
        size_t kt = (size_t)dd * 32 + itile;
        W32[(kt * O_DIM + (o0 + o)) * 16 + kh] = lo | (hi << 16);
    }
}

// ---------------------------------------------------------------------------
// Stage 2: bias[o] = sum_i C[i][o][0]. Grid: O/32 blocks, 256 threads.
// ---------------------------------------------------------------------------
__global__ __launch_bounds__(256) void bias_kernel(
    const float* __restrict__ C, float* __restrict__ bias)
{
    __shared__ float sh[256];
    const int o0 = blockIdx.x * 32;
    const int t  = threadIdx.x;
    const int o  = t & 31;
    const int s  = t >> 5;           // 8 i-slices
    float sum = 0.f;
    for (int i = s; i < I_DIM; i += 8)
        sum += C[((size_t)i * O_DIM + (o0 + o)) * 9];
    sh[t] = sum;
    __syncthreads();
    if (t < 32) {
        float b = 0.f;
#pragma unroll
        for (int s2 = 0; s2 < 8; ++s2) b += sh[s2 * 32 + t];
        bias[o0 + t] = b;
    }
}

// ---------------------------------------------------------------------------
// Stage 3: fused Chebyshev + bf16 WMMA GEMM.
// Grid: (M/128, N/256), 512 threads = 16 wave32 (4 m-groups x 4 n-groups).
// Each wave: 32x64 output = 2x4 accumulators of 16x16 f32.
// LDS: 8 Chebyshev planes of the 128x32 A tile (bf16) = 96 KB.
// ---------------------------------------------------------------------------
__global__ __launch_bounds__(NTHREADS) void cheby_kan_gemm(
    const float* __restrict__ XT,
    const unsigned short* __restrict__ Wb,
    const float* __restrict__ bias,
    float* __restrict__ Y)
{
    __shared__ unsigned short As[8 * BM * LDA];   // 98304 B

    const int t    = threadIdx.x;
    const int wave = t >> 5;
    const int lane = t & 31;
    const int wm   = wave & 3;     // rows wm*32 .. wm*32+31
    const int wn   = wave >> 2;    // cols wn*64 .. wn*64+63
    const int hi   = lane >> 4;    // K-half selector (A/B fragment layout)
    const int ln   = lane & 15;    // row (A) / column (B) within 16
    const int b0   = blockIdx.x * BM;
    const int o0   = blockIdx.y * BN;

    // x-tile staging: thread owns 8 elements: 2 rows x 4 contiguous cols
    const int tcol = (t & 7) * 4;  // 0,4,...,28
    const int trow = t >> 3;       // 0..63 (rows trow, trow+64)

    v8f acc[2][4];
#pragma unroll
    for (int mt = 0; mt < 2; ++mt)
#pragma unroll
        for (int nt = 0; nt < 4; ++nt)
            acc[mt][nt] = (v8f){0.f, 0.f, 0.f, 0.f, 0.f, 0.f, 0.f, 0.f};

    float xt[8], tp[8], tc[8];

    for (int it = 0; it < I_DIM / 32; ++it) {
        // ---- write phase: tanh(x) tile -> all 8 Chebyshev planes in LDS ----
#pragma unroll
        for (int j = 0; j < 2; ++j) {
            int m = trow + 64 * j;
            float4 xv = *(const float4*)&XT[(size_t)(b0 + m) * I_DIM + it * 32 + tcol];
            xt[4 * j + 0] = xv.x;
            xt[4 * j + 1] = xv.y;
            xt[4 * j + 2] = xv.z;
            xt[4 * j + 3] = xv.w;
        }
        // d = 1 plane: T1 = xt
#pragma unroll
        for (int j = 0; j < 2; ++j) {
            int m = trow + 64 * j;
            uint2 packed;
            packed.x = pack2bf(xt[4 * j + 0], xt[4 * j + 1]);
            packed.y = pack2bf(xt[4 * j + 2], xt[4 * j + 3]);
            *(uint2*)&As[m * LDA + tcol] = packed;
        }
#pragma unroll
        for (int e = 0; e < 8; ++e) { tp[e] = 1.0f; tc[e] = xt[e]; }
        // d = 2..8 planes via recurrence
#pragma unroll
        for (int d = 2; d <= 8; ++d) {
#pragma unroll
            for (int j = 0; j < 2; ++j) {
                int m = trow + 64 * j;
                float nv[4];
#pragma unroll
                for (int c = 0; c < 4; ++c) {
                    int e = 4 * j + c;
                    nv[c] = 2.0f * xt[e] * tc[e] - tp[e];
                    tp[e] = tc[e];
                    tc[e] = nv[c];
                }
                uint2 packed;
                packed.x = pack2bf(nv[0], nv[1]);
                packed.y = pack2bf(nv[2], nv[3]);
                *(uint2*)&As[((d - 1) * BM + m) * LDA + tcol] = packed;
            }
        }
        __syncthreads();   // planes visible to all waves

        // ---- read phase: 8 d-steps, 64 WMMAs, no barriers ----
#pragma unroll
        for (int dd = 0; dd < 8; ++dd) {
            const int kt = dd * 32 + it;

            // A fragments from LDS (16-bit A 16x32 layout).
            FragBF a[2];
#pragma unroll
            for (int mt = 0; mt < 2; ++mt) {
                int row = wm * 32 + mt * 16 + ln;
                const unsigned short* p = &As[(dd * BM + row) * LDA + hi * 8];
                a[mt].q[0] = *(const uint4*)p;
                a[mt].q[1] = *(const uint4*)(p + 16);
            }

            // B fragments straight from L2-resident packed W (contiguous tile).
#pragma unroll
            for (int nt = 0; nt < 4; ++nt) {
                FragBF b;
                size_t nbase = ((size_t)kt * O_DIM + (o0 + wn * 64 + nt * 16 + ln)) * 32
                               + (size_t)hi * 8;
                b.q[0] = *(const uint4*)&Wb[nbase];
                b.q[1] = *(const uint4*)&Wb[nbase + 16];
#pragma unroll
                for (int mt = 0; mt < 2; ++mt) {
                    acc[mt][nt] = __builtin_amdgcn_wmma_f32_16x16x32_bf16(
                        false, a[mt].v, false, b.v, (short)0, acc[mt][nt],
                        false, false);
                }
            }
        }
        __syncthreads();   // readers done before next it overwrites planes
    }

    // Epilogue: add T0 bias, store f32 C/D layout
    // (VGPR r: lanes 0-15 -> M=r, lanes 16-31 -> M=r+8).
#pragma unroll
    for (int mt = 0; mt < 2; ++mt) {
#pragma unroll
        for (int nt = 0; nt < 4; ++nt) {
            int mbase = b0 + wm * 32 + mt * 16 + hi * 8;
            int n = o0 + wn * 64 + nt * 16 + ln;
            float bv = bias[n];
#pragma unroll
            for (int r = 0; r < 8; ++r) {
                Y[(size_t)(mbase + r) * O_DIM + n] = acc[mt][nt][r] + bv;
            }
        }
    }
}

// ---------------------------------------------------------------------------
extern "C" void kernel_launch(void* const* d_in, const int* in_sizes, int n_in,
                              void* d_out, int out_size, void* d_ws, size_t ws_size,
                              hipStream_t stream)
{
    const float* x = (const float*)d_in[0];          // (8192, 1024) f32
    const float* c = (const float*)d_in[1];          // (1024, 1024, 9) f32
    float* y = (float*)d_out;                        // (8192, 1024) f32

    const int M  = in_sizes[0] / I_DIM;              // 8192
    const int n4 = in_sizes[0] / 4;

    // Workspace layout: [xt f32 M*I][Wb bf16 8*I*O][bias f32 O]
    float* xt = (float*)d_ws;
    unsigned short* wb = (unsigned short*)((char*)d_ws + (size_t)M * I_DIM * 4);
    float* bias = (float*)((char*)wb + (size_t)8 * I_DIM * O_DIM * 2);

    tanh_kernel<<<(n4 + 255) / 256, 256, 0, stream>>>(
        (const float4*)x, (float4*)xt, n4);

    dim3 pgrid(I_DIM / 32, O_DIM / 32);              // 32 x 32
    cheby_pack_w<<<pgrid, 256, 0, stream>>>(c, wb);

    bias_kernel<<<O_DIM / 32, 256, 0, stream>>>(c, bias);

    dim3 ggrid(M / BM, O_DIM / BN);                  // 64 x 4
    cheby_kan_gemm<<<ggrid, NTHREADS, 0, stream>>>(xt, wb, bias, y);
}
